// Decoder_74363063763402
// MI455X (gfx1250) — compile-verified, matrix operand order checked by reading, not run on hardware
//
#include <hip/hip_runtime.h>
#include <cmath>

// ---------------------------------------------------------------------------
// Problem constants (from the reference)
// ---------------------------------------------------------------------------
constexpr int B_  = 32;
constexpr int T_  = 16;
constexpr int S_  = 1024;
constexpr int H_  = 512;
constexpr int E_  = 512;
constexpr int NH_ = 8;
constexpr int V_  = 50000;
constexpr int OOV_ = 64;
constexpr int I_  = 2048;
constexpr int H2_ = 1024;   // 2*H

typedef __attribute__((ext_vector_type(16))) _Float16 v16h;
typedef __attribute__((ext_vector_type(8)))  float    v8f;

// Pack 8 consecutive fp32 (two float4) into fragment halves [base..base+7].
__device__ __forceinline__ void pack8(v16h& f, int base, float4 a, float4 b)
{
    f[base + 0] = (_Float16)a.x; f[base + 1] = (_Float16)a.y;
    f[base + 2] = (_Float16)a.z; f[base + 3] = (_Float16)a.w;
    f[base + 4] = (_Float16)b.x; f[base + 5] = (_Float16)b.y;
    f[base + 6] = (_Float16)b.z; f[base + 7] = (_Float16)b.w;
}

// Load one 16-wide f16 fragment for this lane.
// f16 16x32 A-layout (ISA 7.12.2): for lane-half hi, regs 0..3 hold
// K = koff+8*hi+{0..7}, regs 4..7 hold K = koff+16+8*hi+{0..7} — i.e. two
// aligned contiguous 8-float chunks -> four b128 loads, branch-free.
__device__ __forceinline__ v16h load_frag(const float* __restrict__ row, int koff, int hi)
{
    const float* p = row + koff + 8 * hi;
    float4 lo0 = *(const float4*)(p);
    float4 lo1 = *(const float4*)(p + 4);
    float4 hi0 = *(const float4*)(p + 16);
    float4 hi1 = *(const float4*)(p + 20);
    v16h f;
    pack8(f, 0, lo0, lo1);
    pack8(f, 8, hi0, hi1);
    return f;
}

// ---------------------------------------------------------------------------
// Generic WMMA GEMM:  C[M,N] = act( A[M,K] @ W[N,K]^T + bias + resid + Cin )
//   A row-major, row stride lda; W torch-style (out,in) so x @ W.T.
//   256 threads = 8 waves, all along N; macro tile 32(M) x 128(N).
//   Each wave: one B fragment reused by two WMMAs (two 16-row M tiles),
//   fp32->f16 convert at load, fp32 accumulate.
//   Requires K % 32 == 0 (true for every matmul here). Out-of-range rows/cols
//   are pointer-clamped: clamped data only reaches C entries never stored.
// act: 0 = none, 1 = exact GELU, 2 = ReLU
// ---------------------------------------------------------------------------
__global__ __launch_bounds__(256)
void gemm_wmma_kernel(const float* __restrict__ A, int lda,
                      const float* __restrict__ W,
                      const float* __restrict__ bias,
                      const float* __restrict__ resid, int ldr,
                      const float* __restrict__ Cin,
                      float* __restrict__ C, int ldc,
                      int M, int N, int K, int act)
{
    const int tid  = threadIdx.x;
    const int wave = tid >> 5;
    const int lane = tid & 31;
    const int ml   = lane & 15;
    const int hi   = lane >> 4;

    const int row0 = blockIdx.y * 32;                  // two 16-row tiles
    const int col0 = blockIdx.x * 128 + wave * 16;     // 16 cols per wave

    // Clamped source rows (finite garbage only pollutes unstored C entries).
    const int ar0 = min(row0 + ml,      M - 1);
    const int ar1 = min(row0 + 16 + ml, M - 1);
    const int bc  = min(col0 + ml,      N - 1);
    const float* __restrict__ A0 = A + (long)ar0 * lda;
    const float* __restrict__ A1 = A + (long)ar1 * lda;
    const float* __restrict__ Wr = W + (long)bc * K;

    v8f acc0 = {}, acc1 = {};

    for (int k0 = 0; k0 < K; k0 += 32) {
        const v16h bf  = load_frag(Wr, k0, hi);
        const v16h af0 = load_frag(A0, k0, hi);
        const v16h af1 = load_frag(A1, k0, hi);
        acc0 = __builtin_amdgcn_wmma_f32_16x16x32_f16(false, af0, false, bf,
                                                      (short)0, acc0, false, false);
        acc1 = __builtin_amdgcn_wmma_f32_16x16x32_f16(false, af1, false, bf,
                                                      (short)0, acc1, false, false);
    }

    const int n = col0 + ml;
    if (n >= N) return;
    const float bsv = bias ? bias[n] : 0.0f;

    #pragma unroll
    for (int r = 0; r < 8; ++r) {
        const int m = row0 + r + (hi << 3);            // C layout: M = r + 8*hi
        #pragma unroll
        for (int t = 0; t < 2; ++t) {
            const int mm = m + 16 * t;
            if (mm < M) {
                float v = (t == 0 ? acc0[r] : acc1[r]) + bsv;
                if (resid) v += resid[(long)mm * ldr + n];
                if (Cin)   v += Cin[(long)mm * ldc + n];
                if (act == 1)      v = 0.5f * v * (1.0f + erff(v * 0.70710678f));
                else if (act == 2) v = fmaxf(v, 0.0f);
                C[(long)mm * ldc + n] = v;
            }
        }
    }
}

// ---------------------------------------------------------------------------
// Last-token multi-head self attention: one block per (b, head), 64 threads.
// ---------------------------------------------------------------------------
__global__ __launch_bounds__(64)
void attn_last_kernel(const float* __restrict__ q,
                      const float* __restrict__ kall,
                      const float* __restrict__ vall,
                      const float* __restrict__ dmask,
                      float* __restrict__ ctx)
{
    const int b = blockIdx.x / NH_;
    const int h = blockIdx.x % NH_;
    const int hd = H_ / NH_;          // 64
    __shared__ float sc[T_];
    const int tid = threadIdx.x;

    if (tid < T_) {
        const float* qp = q + (long)b * H_ + h * hd;
        const float* kp = kall + ((long)b * T_ + tid) * H_ + h * hd;
        float s = 0.f;
        for (int d = 0; d < hd; ++d) s += qp[d] * kp[d];
        s *= 0.125f;                                    // 1/sqrt(64)
        s += (1.0f - dmask[(long)b * T_ + tid]) * -10000.0f;
        sc[tid] = s;
    }
    __syncthreads();
    if (tid == 0) {
        float mx = sc[0];
        for (int j = 1; j < T_; ++j) mx = fmaxf(mx, sc[j]);
        float sum = 0.f;
        for (int j = 0; j < T_; ++j) { float e = __expf(sc[j] - mx); sc[j] = e; sum += e; }
        const float inv = 1.0f / sum;
        for (int j = 0; j < T_; ++j) sc[j] *= inv;
    }
    __syncthreads();
    float o = 0.f;
    for (int j = 0; j < T_; ++j)
        o += sc[j] * vall[((long)b * T_ + j) * H_ + h * hd + tid];
    ctx[(long)b * H_ + h * hd + tid] = o;
}

// ---------------------------------------------------------------------------
// Row layernorm (width Wd), optional row-mask (zero row if mask==0).
// ---------------------------------------------------------------------------
__global__ __launch_bounds__(256)
void layernorm_kernel(const float* __restrict__ X, float* __restrict__ Y,
                      const float* __restrict__ g, const float* __restrict__ bb,
                      const float* __restrict__ mask, int mstride, int Wd)
{
    const int row = blockIdx.x;
    const int tid = threadIdx.x;
    __shared__ float red[256];
    const float* xr = X + (long)row * Wd;

    float s = 0.f;
    for (int c = tid; c < Wd; c += 256) s += xr[c];
    red[tid] = s; __syncthreads();
    for (int off = 128; off > 0; off >>= 1) { if (tid < off) red[tid] += red[tid + off]; __syncthreads(); }
    const float mean = red[0] / Wd;
    __syncthreads();

    float vs = 0.f;
    for (int c = tid; c < Wd; c += 256) { const float d = xr[c] - mean; vs += d * d; }
    red[tid] = vs; __syncthreads();
    for (int off = 128; off > 0; off >>= 1) { if (tid < off) red[tid] += red[tid + off]; __syncthreads(); }
    const float inv = rsqrtf(red[0] / Wd + 1e-12f);

    float mk = 1.0f;
    if (mask) mk = (mask[(long)row * mstride] == 0.0f) ? 0.0f : 1.0f;
    for (int c = tid; c < Wd; c += 256)
        Y[(long)row * Wd + c] = ((xr[c] - mean) * inv * g[c] + bb[c]) * mk;
}

// ---------------------------------------------------------------------------
// Row-wise concat of two matrices.
// ---------------------------------------------------------------------------
__global__ void concat_kernel(const float* __restrict__ s1, int ld1, int w1,
                              const float* __restrict__ s2, int ld2, int w2,
                              float* __restrict__ dst, int rows)
{
    const int wt = w1 + w2;
    const int i = blockIdx.x * blockDim.x + threadIdx.x;
    if (i >= rows * wt) return;
    const int r = i / wt, c = i % wt;
    dst[i] = (c < w1) ? s1[(long)r * ld1 + c] : s2[(long)r * ld2 + (c - w1)];
}

// ---------------------------------------------------------------------------
// LSTM cell elementwise; also writes status = [h, c].
// ---------------------------------------------------------------------------
__global__ void lstm_kernel(const float* __restrict__ gates,
                            const float* __restrict__ c0,
                            float* __restrict__ h, float* __restrict__ status)
{
    const int i = blockIdx.x * blockDim.x + threadIdx.x;
    if (i >= B_ * H_) return;
    const int b = i / H_, j = i % H_;
    const float* g = gates + (long)b * 4 * H_;
    const float ig = g[j], fg = g[H_ + j], gg = g[2 * H_ + j], og = g[3 * H_ + j];
    const float sf = 1.f / (1.f + __expf(-fg));
    const float si = 1.f / (1.f + __expf(-ig));
    const float so = 1.f / (1.f + __expf(-og));
    const float c  = sf * c0[i] + si * tanhf(gg);
    const float hh = so * tanhf(c);
    h[i] = hh;
    status[(long)b * H2_ + j]       = hh;
    status[(long)b * H2_ + H_ + j]  = c;
}

// ---------------------------------------------------------------------------
// Encoder attention scores: e[b,s] = av . tanh(feat + dec_feat + cov_feat)
// One wave per s-row (coalesced row read + shuffle reduce).
// ---------------------------------------------------------------------------
__global__ __launch_bounds__(256)
void enc_score_kernel(const float* __restrict__ feat,
                      const float* __restrict__ decfeat,
                      const float* __restrict__ cov,
                      const float* __restrict__ aw_c,
                      const float* __restrict__ ab_c,
                      const float* __restrict__ av,
                      const float* __restrict__ emask,
                      float* __restrict__ e)
{
    const int b    = blockIdx.y;
    const int wave = threadIdx.x >> 5;
    const int lane = threadIdx.x & 31;
    const int s    = blockIdx.x * 8 + wave;

    const float* fr = feat + ((long)b * S_ + s) * H2_;
    const float* df = decfeat + (long)b * H2_;
    const float cv  = cov[(long)b * S_ + s];

    float sum = 0.f;
    for (int c = lane; c < H2_; c += 32)
        sum += tanhf(fr[c] + df[c] + cv * aw_c[c] + ab_c[c]) * av[c];
    for (int off = 16; off > 0; off >>= 1) sum += __shfl_xor(sum, off, 32);

    if (lane == 0) {
        e[(long)b * S_ + s] =
            (emask[(long)b * S_ + s] == 0.0f) ? -__builtin_inff() : sum;
    }
}

// ---------------------------------------------------------------------------
// Softmax over n elements per row (grid = rows).
// ---------------------------------------------------------------------------
__global__ __launch_bounds__(256)
void softmax_1d_kernel(const float* __restrict__ e, float* __restrict__ a, int n)
{
    const int b = blockIdx.x, tid = threadIdx.x;
    __shared__ float red[256];
    const float* er = e + (long)b * n;

    float mx = -__builtin_inff();
    for (int i = tid; i < n; i += 256) mx = fmaxf(mx, er[i]);
    red[tid] = mx; __syncthreads();
    for (int off = 128; off > 0; off >>= 1) { if (tid < off) red[tid] = fmaxf(red[tid], red[tid + off]); __syncthreads(); }
    mx = red[0]; __syncthreads();

    float sm = 0.f;
    for (int i = tid; i < n; i += 256) sm += __expf(er[i] - mx);
    red[tid] = sm; __syncthreads();
    for (int off = 128; off > 0; off >>= 1) { if (tid < off) red[tid] += red[tid + off]; __syncthreads(); }
    const float inv = 1.0f / red[0];

    for (int i = tid; i < n; i += 256) a[(long)b * n + i] = __expf(er[i] - mx) * inv;
}

// ---------------------------------------------------------------------------
// cur_ctx[b,c] = sum_s a[b,s] * enc_out[b,s,c] — coalesced over c,
// with speculative prefetch of upcoming rows.
// ---------------------------------------------------------------------------
__global__ __launch_bounds__(256)
void enc_ctx_kernel(const float* __restrict__ a, const float* __restrict__ eo,
                    float* __restrict__ cc)
{
    const int b = blockIdx.y;
    const int c = blockIdx.x * 256 + threadIdx.x;
    const float* ar = a + (long)b * S_;
    const float* er = eo + (long)b * S_ * H2_ + c;
    float acc = 0.f;
    for (int s = 0; s < S_; ++s) {
        if ((s & 63) == 0 && s + 64 < S_)
            __builtin_prefetch(er + (long)(s + 64) * H2_, 0, 1);
        acc += ar[s] * er[(long)s * H2_];
    }
    cc[(long)b * H2_ + c] = acc;
}

// ---------------------------------------------------------------------------
// Pointer-gate: gp[b] = sigmoid(cc.gw_h + status.gw_s + x.gw_x + biases)
// ---------------------------------------------------------------------------
__global__ __launch_bounds__(256)
void gp_kernel(const float* __restrict__ cc, const float* __restrict__ status,
               const float* __restrict__ x,
               const float* __restrict__ gw_h, const float* __restrict__ gb_h,
               const float* __restrict__ gw_s, const float* __restrict__ gb_s,
               const float* __restrict__ gw_x, const float* __restrict__ gb_x,
               float* __restrict__ gp)
{
    const int b = blockIdx.x, tid = threadIdx.x;
    __shared__ float red[256];
    float s = 0.f;
    for (int c = tid; c < H2_; c += 256)
        s += cc[(long)b * H2_ + c] * gw_h[c] + status[(long)b * H2_ + c] * gw_s[c];
    for (int c = tid; c < H_; c += 256)
        s += x[(long)b * H_ + c] * gw_x[c];
    red[tid] = s; __syncthreads();
    for (int off = 128; off > 0; off >>= 1) { if (tid < off) red[tid] += red[tid + off]; __syncthreads(); }
    if (tid == 0) {
        const float t = red[0] + gb_h[0] + gb_s[0] + gb_x[0];
        gp[b] = 1.f / (1.f + __expf(-t));
    }
}

// ---------------------------------------------------------------------------
// Vocab softmax * gp, and zero the OOV extension region.
// ---------------------------------------------------------------------------
__global__ __launch_bounds__(256)
void vocab_final_kernel(const float* __restrict__ logits,
                        const float* __restrict__ gp,
                        float* __restrict__ out)
{
    const int b = blockIdx.x, tid = threadIdx.x;
    __shared__ float red[256];
    const float* lr = logits + (long)b * V_;

    float mx = -__builtin_inff();
    for (int i = tid; i < V_; i += 256) mx = fmaxf(mx, lr[i]);
    red[tid] = mx; __syncthreads();
    for (int off = 128; off > 0; off >>= 1) { if (tid < off) red[tid] = fmaxf(red[tid], red[tid + off]); __syncthreads(); }
    mx = red[0]; __syncthreads();

    float sm = 0.f;
    for (int i = tid; i < V_; i += 256) sm += __expf(lr[i] - mx);
    red[tid] = sm; __syncthreads();
    for (int off = 128; off > 0; off >>= 1) { if (tid < off) red[tid] += red[tid + off]; __syncthreads(); }
    const float inv = 1.0f / red[0];
    const float g = gp[b];

    float* orow = out + (long)b * (V_ + OOV_);
    for (int i = tid; i < V_; i += 256) orow[i] = __expf(lr[i] - mx) * inv * g;
    for (int i = V_ + tid; i < V_ + OOV_; i += 256) orow[i] = 0.f;
}

// ---------------------------------------------------------------------------
// Copy-distribution scatter-add.
// ---------------------------------------------------------------------------
__global__ void scatter_kernel(const float* __restrict__ a,
                               const float* __restrict__ gp,
                               const int* __restrict__ idx,
                               float* __restrict__ out)
{
    const int i = blockIdx.x * blockDim.x + threadIdx.x;
    if (i >= B_ * S_) return;
    const int b = i / S_;
    atomicAdd(&out[(long)b * (V_ + OOV_) + idx[i]], a[i] * (1.0f - gp[b]));
}

// ---------------------------------------------------------------------------
// Host launcher
// ---------------------------------------------------------------------------
extern "C" void kernel_launch(void* const* d_in, const int* in_sizes, int n_in,
                              void* d_out, int out_size, void* d_ws, size_t ws_size,
                              hipStream_t stream)
{
    (void)in_sizes; (void)n_in; (void)out_size; (void)ws_size;
    // ---- inputs (setup_inputs order, params dict flattened in insertion order)
    const float* x_in     = (const float*)d_in[0];   // (B,T,E)
    const float* dmask    = (const float*)d_in[1];   // (B,T)
    const float* h0       = (const float*)d_in[2];   // (1,B,H)
    const float* c0       = (const float*)d_in[3];   // (1,B,H)
    const float* enc_out  = (const float*)d_in[4];   // (B,S,H2)
    const float* enc_feat = (const float*)d_in[5];   // (B,S,H2)
    const float* enc_mask = (const float*)d_in[6];   // (B,S)
    const float* ctxvec   = (const float*)d_in[7];   // (B,H2)
    const int*   enc_oov  = (const int*)d_in[9];     // (B,S)
    const float* coverage = (const float*)d_in[10];  // (B,S)
    int p = 12;
    const float* Wq    = (const float*)d_in[p++]; const float* bq    = (const float*)d_in[p++];
    const float* Wk    = (const float*)d_in[p++]; const float* bk    = (const float*)d_in[p++];
    const float* Wv    = (const float*)d_in[p++]; const float* bv    = (const float*)d_in[p++];
    const float* Wo    = (const float*)d_in[p++]; const float* bo    = (const float*)d_in[p++];
    const float* ln1_g = (const float*)d_in[p++]; const float* ln1_b = (const float*)d_in[p++];
    const float* Wi    = (const float*)d_in[p++]; const float* bi    = (const float*)d_in[p++];
    const float* Wout  = (const float*)d_in[p++]; const float* bout  = (const float*)d_in[p++];
    const float* ln2_g = (const float*)d_in[p++]; const float* ln2_b = (const float*)d_in[p++];
    const float* gli_W = (const float*)d_in[p++]; const float* gli_b = (const float*)d_in[p++];
    const float* W_ih  = (const float*)d_in[p++]; const float* W_hh  = (const float*)d_in[p++];
    const float* b_ih  = (const float*)d_in[p++]; const float* b_hh  = (const float*)d_in[p++];
    const float* aw_s  = (const float*)d_in[p++]; const float* aw_c  = (const float*)d_in[p++];
    const float* ab_c  = (const float*)d_in[p++]; const float* av    = (const float*)d_in[p++];
    const float* gw_h  = (const float*)d_in[p++]; const float* gb_h  = (const float*)d_in[p++];
    const float* gw_s  = (const float*)d_in[p++]; const float* gb_s  = (const float*)d_in[p++];
    const float* gw_x  = (const float*)d_in[p++]; const float* gb_x  = (const float*)d_in[p++];
    const float* oW1   = (const float*)d_in[p++]; const float* ob1   = (const float*)d_in[p++];
    const float* oW2   = (const float*)d_in[p++]; const float* ob2   = (const float*)d_in[p++];

    // ---- workspace layout (floats; every block multiple of 16 floats -> 64B aligned)
    float* ws = (float*)d_ws;
    float* ws_k       = ws;                                // B*T*H
    float* ws_v       = ws_k      + (size_t)B_ * T_ * H_;  // B*T*H
    float* ws_q       = ws_v      + (size_t)B_ * T_ * H_;  // B*H
    float* ws_ctx     = ws_q      + (size_t)B_ * H_;
    float* ws_tmp     = ws_ctx    + (size_t)B_ * H_;
    float* ws_attn    = ws_tmp    + (size_t)B_ * H_;
    float* ws_inter   = ws_attn   + (size_t)B_ * H_;       // B*I
    float* ws_bert    = ws_inter  + (size_t)B_ * I_;
    float* ws_cat     = ws_bert   + (size_t)B_ * H_;       // B*(H2+E)
    float* ws_x       = ws_cat    + (size_t)B_ * (H2_ + E_);
    float* ws_gates   = ws_x      + (size_t)B_ * E_;       // B*4H
    float* ws_h       = ws_gates  + (size_t)B_ * 4 * H_;
    float* ws_status  = ws_h      + (size_t)B_ * H_;       // B*H2
    float* ws_decfeat = ws_status + (size_t)B_ * H2_;
    float* ws_e       = ws_decfeat+ (size_t)B_ * H2_;      // B*S
    float* ws_a       = ws_e      + (size_t)B_ * S_;
    float* ws_cc      = ws_a      + (size_t)B_ * S_;       // B*H2
    float* ws_gp      = ws_cc     + (size_t)B_ * H2_;      // B (padded to 64)
    float* ws_of      = ws_gp     + 64;                    // B*(H+H2)
    float* ws_relu    = ws_of     + (size_t)B_ * (H_ + H2_);
    float* ws_logits  = ws_relu   + (size_t)B_ * H_;       // B*V

    float* out = (float*)d_out;

    auto gemm = [&](const float* A, int lda, const float* W, const float* bias,
                    const float* resid, int ldr, const float* Cin, float* C, int ldc,
                    int M, int N, int K, int act) {
        dim3 grid((N + 127) / 128, (M + 31) / 32);
        gemm_wmma_kernel<<<grid, 256, 0, stream>>>(A, lda, W, bias, resid, ldr,
                                                   Cin, C, ldc, M, N, K, act);
    };

    const float* x_last = x_in + (size_t)(T_ - 1) * H_;    // row stride T*H -> one row per batch

    // ---- transformer block (K/V for all tokens; Q/attn/FFN only for last token)
    gemm(x_in,   H_,      Wk, bk, nullptr, 0, nullptr, ws_k, H_, B_ * T_, H_, H_, 0);
    gemm(x_in,   H_,      Wv, bv, nullptr, 0, nullptr, ws_v, H_, B_ * T_, H_, H_, 0);
    gemm(x_last, T_ * H_, Wq, bq, nullptr, 0, nullptr, ws_q, H_, B_,      H_, H_, 0);
    attn_last_kernel<<<B_ * NH_, 64, 0, stream>>>(ws_q, ws_k, ws_v, dmask, ws_ctx);
    gemm(ws_ctx, H_, Wo, bo, x_last, T_ * H_, nullptr, ws_tmp, H_, B_, H_, H_, 0);
    layernorm_kernel<<<B_, 256, 0, stream>>>(ws_tmp, ws_attn, ln1_g, ln1_b, nullptr, 0, H_);
    gemm(ws_attn, H_, Wi, bi, nullptr, 0, nullptr, ws_inter, I_, B_, I_, H_, 1 /*gelu*/);
    gemm(ws_inter, I_, Wout, bout, ws_attn, H_, nullptr, ws_tmp, H_, B_, H_, I_, 0);
    layernorm_kernel<<<B_, 256, 0, stream>>>(ws_tmp, ws_bert, ln2_g, ln2_b,
                                             dmask + (T_ - 1), T_, H_);

    // ---- gli projection + LSTM cell
    {
        int total = B_ * (H2_ + H_);
        concat_kernel<<<(total + 255) / 256, 256, 0, stream>>>(ctxvec, H2_, H2_,
                                                               ws_bert, H_, H_, ws_cat, B_);
    }
    gemm(ws_cat, H2_ + E_, gli_W, gli_b, nullptr, 0, nullptr, ws_x, E_, B_, E_, H2_ + E_, 0);
    gemm(ws_x, E_, W_ih, b_ih, nullptr, 0, nullptr,  ws_gates, 4 * H_, B_, 4 * H_, E_, 0);
    gemm(h0,   H_, W_hh, b_hh, nullptr, 0, ws_gates, ws_gates, 4 * H_, B_, 4 * H_, H_, 0);
    lstm_kernel<<<(B_ * H_ + 255) / 256, 256, 0, stream>>>(ws_gates, c0, ws_h, ws_status);

    // ---- encoder attention (bandwidth-critical: coalesced streaming)
    gemm(ws_status, H2_, aw_s, nullptr, nullptr, 0, nullptr, ws_decfeat, H2_, B_, H2_, H2_, 0);
    enc_score_kernel<<<dim3(S_ / 8, B_), 256, 0, stream>>>(enc_feat, ws_decfeat, coverage,
                                                           aw_c, ab_c, av, enc_mask, ws_e);
    softmax_1d_kernel<<<B_, 256, 0, stream>>>(ws_e, ws_a, S_);
    enc_ctx_kernel<<<dim3(H2_ / 256, B_), 256, 0, stream>>>(ws_a, enc_out, ws_cc);
    gp_kernel<<<B_, 256, 0, stream>>>(ws_cc, ws_status, ws_x,
                                      gw_h, gb_h, gw_s, gb_s, gw_x, gb_x, ws_gp);

    // ---- output projection + pointer-generator mix
    {
        int total = B_ * (H_ + H2_);
        concat_kernel<<<(total + 255) / 256, 256, 0, stream>>>(ws_h, H_, H_,
                                                               ws_cc, H2_, H2_, ws_of, B_);
    }
    gemm(ws_of,   H_ + H2_, oW1, ob1, nullptr, 0, nullptr, ws_relu,   H_, B_, H_, H_ + H2_, 2 /*relu*/);
    gemm(ws_relu, H_,       oW2, ob2, nullptr, 0, nullptr, ws_logits, V_, B_, V_, H_,       0);
    vocab_final_kernel<<<B_, 256, 0, stream>>>(ws_logits, ws_gp, out);
    scatter_kernel<<<(B_ * S_ + 255) / 256, 256, 0, stream>>>(ws_a, ws_gp, enc_oov, out);
}
